// RotaryModel_10977936408971
// MI455X (gfx1250) — matrix operand order repaired and checked by reading, not compiled, not run on hardware
//
#include <hip/hip_runtime.h>

typedef __attribute__((ext_vector_type(16))) _Float16 v16h;
typedef __attribute__((ext_vector_type(8)))  float    v8f;

#define BATCH   4
#define SEQ     4096
#define HID     2048
#define HALF_H  1024
#define M_TOTAL (BATCH * SEQ)   // 16384
#define BM      128             // M rows per block
#define BN_HALF 64              // lo columns per block (paired with +1024)
#define BK      32              // K step (one 16x16x32 WMMA per step)
#define LDS_STRIDE 56           // halves per row = 112 B = 28 banks (conflict-free b128)

#if __has_builtin(__builtin_amdgcn_global_load_async_to_lds_b128)
#define HAS_ASYNC_LDS 1
typedef int v4i_b128 __attribute__((vector_size(16)));
typedef __attribute__((address_space(1))) v4i_b128* g128_t;
typedef __attribute__((address_space(3))) v4i_b128* l128_t;
#else
#define HAS_ASYNC_LDS 0
#endif

__device__ __forceinline__ void stage_wait_async() {
#if HAS_ASYNC_LDS
#if __has_builtin(__builtin_amdgcn_s_wait_asynccnt)
    __builtin_amdgcn_s_wait_asynccnt(0);
#else
    asm volatile("s_wait_asynccnt 0x0" ::: "memory");
#endif
#endif
}

__device__ __forceinline__ void sched_fence() {
#if __has_builtin(__builtin_amdgcn_sched_barrier)
    __builtin_amdgcn_sched_barrier(0);
#endif
}

__device__ __forceinline__ void copy16_to_lds(const _Float16* g, _Float16* l) {
#if HAS_ASYNC_LDS
    __builtin_amdgcn_global_load_async_to_lds_b128(
        (g128_t)(unsigned long long)(uintptr_t)g,
        (l128_t)(unsigned int)(uintptr_t)l, 0, 0);
#else
    *(uint4*)l = *(const uint4*)g;
#endif
}

// ---------------------------------------------------------------------------
// fp32 -> fp16 pre-conversion (x and W), 8 elements / thread, b128 in & out
// ---------------------------------------------------------------------------
__global__ __launch_bounds__(256)
void cvt_f32_f16_kernel(const float* __restrict__ in, _Float16* __restrict__ out,
                        long long n)
{
    long long i = (long long)blockIdx.x * blockDim.x + threadIdx.x;
    if (i * 8 + 8 > n) return;
    float4 a = ((const float4*)in)[i * 2];
    float4 b = ((const float4*)in)[i * 2 + 1];
    union { _Float16 h[8]; uint4 u; } r;
    r.h[0] = (_Float16)a.x; r.h[1] = (_Float16)a.y;
    r.h[2] = (_Float16)a.z; r.h[3] = (_Float16)a.w;
    r.h[4] = (_Float16)b.x; r.h[5] = (_Float16)b.y;
    r.h[6] = (_Float16)b.z; r.h[7] = (_Float16)b.w;
    ((uint4*)out)[i] = r.u;
}

// ---------------------------------------------------------------------------
// Fused GEMM (q = x @ W^T + b) + RoPE + 2x scale + fp16 store.
// Block: 256 threads = 8 waves, tile 128(M) x {64 lo cols + 64 hi cols}.
// Wave (wm, wn): rows [wm*64,+64), cols {n0+wn*16..+16} and partners {+1024}.
// ---------------------------------------------------------------------------
__global__ __launch_bounds__(256)
void rope_gemm_kernel(const _Float16* __restrict__ xh,
                      const _Float16* __restrict__ wh,
                      const float*    __restrict__ bias,
                      _Float16*       __restrict__ out)
{
    __shared__ _Float16 As[2][BM][LDS_STRIDE];  // 28 KB
    __shared__ _Float16 Bs[2][BM][LDS_STRIDE];  // 28 KB; rows 0..63 lo, 64..127 hi

    const int tid  = threadIdx.x;
    const int lane = tid & 31;
    const int wid  = tid >> 5;
    const int wm   = wid >> 2;      // 0..1 : M half of the block tile
    const int wn   = wid & 3;       // 0..3 : 16-col group
    const int lrow = lane & 15;
    const int lhi  = lane >> 4;     // half-wave selector (ISA frag layout)

    const int m0 = blockIdx.y * BM;
    const int n0 = blockIdx.x * BN_HALF;    // lo column base, in [0,1024)

    v8f acc_lo[4] = {};
    v8f acc_hi[4] = {};

    // Persistent per-thread staging pointers, bumped by BK halves per stage.
    const int r = tid >> 2;            // 0..63
    const int c = (tid & 3) * 8;       // half-index 0,8,16,24
    const _Float16* gA0 = xh + (size_t)(m0 + r)          * HID + c;  // A rows 0..63
    const _Float16* gA1 = gA0 + (size_t)64 * HID;                    // A rows 64..127
    const _Float16* gB0 = wh + (size_t)(n0 + r)          * HID + c;  // W lo rows
    const _Float16* gB1 = wh + (size_t)(n0 + HALF_H + r) * HID + c;  // W hi rows

    auto issue_stage = [&](int buf) {
        copy16_to_lds(gA0, &As[buf][r     ][c]);
        copy16_to_lds(gA1, &As[buf][r + 64][c]);
        copy16_to_lds(gB0, &Bs[buf][r     ][c]);
        copy16_to_lds(gB1, &Bs[buf][r + 64][c]);
        gA0 += BK; gA1 += BK; gB0 += BK; gB1 += BK;
    };

    issue_stage(0);
    stage_wait_async();
    __syncthreads();

    for (int k0 = 0; k0 < HID; k0 += BK) {
        const int buf = (k0 >> 5) & 1;
        if (k0 + BK < HID) issue_stage(buf ^ 1);

        // ---- batch ALL fragment ds_loads, then fence, then the 8 WMMAs ----
        // B fragments (32x16): lanes 0-15 hold K 0..15, lanes 16-31 hold K 16..31
        v16h bfrag_lo, bfrag_hi;
        {
            const _Float16* bp = &Bs[buf][wn * 16 + lrow][lhi * 16];
            ((uint4*)&bfrag_lo)[0] = ((const uint4*)bp)[0];
            ((uint4*)&bfrag_lo)[1] = ((const uint4*)bp)[1];
            const _Float16* bq = &Bs[buf][64 + wn * 16 + lrow][lhi * 16];
            ((uint4*)&bfrag_hi)[0] = ((const uint4*)bq)[0];
            ((uint4*)&bfrag_hi)[1] = ((const uint4*)bq)[1];
        }
        // A fragments (16x32): lanes 0-15 hold K 0..15, lanes 16-31 hold K 16..31
        v16h afrag[4];
        #pragma unroll
        for (int i = 0; i < 4; i++) {
            const _Float16* ap = &As[buf][wm * 64 + i * 16 + lrow][lhi * 16];
            ((uint4*)&afrag[i])[0] = *(const uint4*)ap;
            ((uint4*)&afrag[i])[1] = *(const uint4*)(ap + 8);
        }

        sched_fence();   // keep the 12 ds_loads ahead of the WMMA block

        #pragma unroll
        for (int i = 0; i < 4; i++) {
            acc_lo[i] = __builtin_amdgcn_wmma_f32_16x16x32_f16(
                false, afrag[i], false, bfrag_lo, (short)0, acc_lo[i], false, false);
            acc_hi[i] = __builtin_amdgcn_wmma_f32_16x16x32_f16(
                false, afrag[i], false, bfrag_hi, (short)0, acc_hi[i], false, false);
        }

        stage_wait_async();   // async writes to buf^1 done before barrier
        __syncthreads();
    }

    // ---------------- fused bias + RoPE + 2x + fp16 store ----------------
    // C/D layout: VGPR r, lanes 0-15 -> M = r, lanes 16-31 -> M = r + 8; N = lane&15
    const int col = n0 + wn * 16 + lrow;                 // lo column, [0,1024)
    const float inv_freq = __expf((float)col * -0.008994473996f); // -ln(1e4)/1024
    const float b_lo = bias[col];
    const float b_hi = bias[col + HALF_H];

    #pragma unroll
    for (int i = 0; i < 4; i++) {
        const int mbase = m0 + wm * 64 + i * 16 + lhi * 8;
        #pragma unroll
        for (int rr = 0; rr < 8; rr++) {
            const int m = mbase + rr;
            const int s = m & (SEQ - 1);                 // m = b*4096 + s
            float sn, cs;
            __sincosf((float)s * inv_freq, &sn, &cs);
            const float ql = acc_lo[i][rr] + b_lo;
            const float qh = acc_hi[i][rr] + b_hi;
            out[(size_t)m * HID + col]          = (_Float16)(2.0f * (ql * cs - qh * sn));
            out[(size_t)m * HID + col + HALF_H] = (_Float16)(2.0f * (qh * cs + ql * sn));
        }
    }
}

// ---------------------------------------------------------------------------
extern "C" void kernel_launch(void* const* d_in, const int* in_sizes, int n_in,
                              void* d_out, int out_size, void* d_ws, size_t ws_size,
                              hipStream_t stream)
{
    const float* x    = (const float*)d_in[0];   // (4, 4096, 2048) f32
    const float* W    = (const float*)d_in[1];   // (2048, 2048)    f32
    const float* bias = (const float*)d_in[2];   // (2048,)         f32
    _Float16* out = (_Float16*)d_out;            // (4,1,4096,2048) f16

    const long long xN = (long long)M_TOTAL * HID;   // 33,554,432
    const long long wN = (long long)HID * HID;       //  4,194,304
    _Float16* xh = (_Float16*)d_ws;                  // 64 MB
    _Float16* wh = xh + xN;                          //  8 MB (ws total 72 MB)

    cvt_f32_f16_kernel<<<(unsigned)((xN / 8 + 255) / 256), 256, 0, stream>>>(x, xh, xN);
    cvt_f32_f16_kernel<<<(unsigned)((wN / 8 + 255) / 256), 256, 0, stream>>>(W, wh, wN);

    dim3 grid(HALF_H / BN_HALF, M_TOTAL / BM);       // (16, 128) = 2048 blocks
    rope_gemm_kernel<<<grid, 256, 0, stream>>>(xh, wh, bias, out);
}
